// primitiveCLloss_75685913690506
// MI455X (gfx1250) — compile-verified, matrix operand order checked by reading, not run on hardware
//
#include <hip/hip_runtime.h>
#include <hip/hip_bf16.h>
#include <math.h>

#define BS_ 8
#define N_  40000
#define C_  256
#define P_  64
#define K_  2048
// T = 0.2 -> 1/T = 5.0 ; W = 0.1

typedef __attribute__((ext_vector_type(2)))  float    v2f;
typedef __attribute__((ext_vector_type(8)))  float    v8f;
typedef __attribute__((ext_vector_type(16))) _Float16 v16h;

#if defined(__has_builtin)
#  if __has_builtin(__builtin_amdgcn_wmma_f32_16x16x4_f32)
#    define USE_WMMA_F32X4 1
#  endif
#endif
#ifndef USE_WMMA_F32X4
#  define USE_WMMA_F32X4 0
#endif

// ---------------------------------------------------------------------------
// Kernel 1: per-(n,b) inverse L2 norm. One wave32 per 256-float row.
// grid = N*BS/8 blocks of 256 threads (8 waves).
// ---------------------------------------------------------------------------
__global__ void k_invnorm(const float* __restrict__ features,
                          float* __restrict__ invnorm) {
    const int wave = threadIdx.x >> 5;
    const int lane = threadIdx.x & 31;
    const int r = blockIdx.x * 8 + wave;        // r = n*BS + b  (< 320000)
    const float4* f4 = (const float4*)(features + (size_t)r * C_ + lane * 8);
    float4 x = f4[0], y = f4[1];
    float s = x.x*x.x + x.y*x.y + x.z*x.z + x.w*x.w
            + y.x*y.x + y.y*y.y + y.z*y.z + y.w*y.w;
    #pragma unroll
    for (int off = 16; off > 0; off >>= 1) s += __shfl_xor(s, off, 32);
    if (lane == 0) invnorm[r] = rsqrtf(s);
}

// ---------------------------------------------------------------------------
// Kernel 2: ragged gather-accumulate. One block per (b,p); thread = channel c.
// Indices staged in LDS; every k-step is a coalesced 1KB row read (L2-resident
// per-batch working set). Writes sums_bp[BS,P,C] and counts_bp[BS,P].
// ---------------------------------------------------------------------------
__global__ void k_gather(const int*   __restrict__ primlabel,
                         const float* __restrict__ features,
                         const float* __restrict__ invnorm,
                         float* __restrict__ sums_bp,
                         float* __restrict__ counts_bp) {
    __shared__ int s_idx[K_];
    __shared__ int s_cnt;
    const int bp = blockIdx.x;          // b*P + p
    const int b  = bp >> 6;             // P_ == 64
    const int c  = threadIdx.x;         // 0..255
    if (c == 0) s_cnt = 0;
    __syncthreads();

    int cnt = 0;
    const int base = bp * K_;
    #pragma unroll
    for (int j = 0; j < K_ / 256; ++j) {
        int t = primlabel[base + j * 256 + c];
        s_idx[j * 256 + c] = t;
        cnt += (t >= 0) ? 1 : 0;
    }
    atomicAdd(&s_cnt, cnt);
    __syncthreads();

    float acc = 0.0f;
    #pragma unroll 4
    for (int k = 0; k < K_; ++k) {
        int t = s_idx[k];
        if (t >= 0) {                           // uniform branch across block
            int row = t * BS_ + b;
            float inv = invnorm[row];           // broadcast load
            acc = fmaf(features[(size_t)row * C_ + c], inv, acc);
        }
    }
    sums_bp[(size_t)bp * C_ + c] = acc;
    if (c == 0) counts_bp[bp] = (float)s_cnt;
}

// ---------------------------------------------------------------------------
// Kernel 3: reduce over b, mean with prototype fallback, L2-normalize.
// One block per p, thread = channel c.
// ---------------------------------------------------------------------------
__global__ void k_mean_norm(const float* __restrict__ sums_bp,
                            const float* __restrict__ counts_bp,
                            const float* __restrict__ prototype,
                            float* __restrict__ prim) {
    __shared__ float red[256];
    const int p = blockIdx.x;
    const int c = threadIdx.x;
    float s = 0.0f, cnt = 0.0f;
    #pragma unroll
    for (int b = 0; b < BS_; ++b) {
        s   += sums_bp[(size_t)(b * P_ + p) * C_ + c];
        cnt += counts_bp[b * P_ + p];
    }
    float mean = (cnt > 0.0f) ? (s / cnt) : prototype[p * C_ + c];
    red[c] = mean * mean;
    __syncthreads();
    #pragma unroll
    for (int off = 128; off > 0; off >>= 1) {
        if (c < off) red[c] += red[c + off];
        __syncthreads();
    }
    prim[p * C_ + c] = mean * rsqrtf(red[0]);
}

// ---------------------------------------------------------------------------
// Kernel 4: sim = prim @ proto^T / T via WMMA, fused InfoNCE loss.
// Single block, 512 threads = 16 waves; wave (ti,tj) owns 16x16 tile of the
// 64x64 sim matrix. K-loop over C=256. EXEC is all-1s (no divergence).
// ---------------------------------------------------------------------------
__global__ void k_sim_loss(const float* __restrict__ prim,
                           const float* __restrict__ prototype,
                           float* __restrict__ out) {
    __shared__ float s_sim[64][65];   // pad: avoid bank conflicts on row reads
    __shared__ float s_nll[64];

    const int tid  = threadIdx.x;
    const int wave = tid >> 5;
    const int lane = tid & 31;
    const int ti = wave >> 2, tj = wave & 3;
    const int mrow = ti * 16 + (lane & 15);   // A row (p)
    const int nrow = tj * 16 + (lane & 15);   // B col row in proto (q)

    v8f acc = {};
#if USE_WMMA_F32X4
    // A 16x4 f32: lane<16 -> K0,K1 ; lane>=16 -> K2,K3. B symmetric.
    const int ksub = (lane >> 4) * 2;
    for (int kk = 0; kk < C_; kk += 4) {
        v2f a = *(const v2f*)(prim      + mrow * C_ + kk + ksub);
        v2f b = *(const v2f*)(prototype + nrow * C_ + kk + ksub);
        acc = __builtin_amdgcn_wmma_f32_16x16x4_f32(
                false, a, false, b, (short)0, acc, false, false);
    }
#else
    // f16 fallback (inputs are unit-normalized rows; |x| <= 1, f16-safe).
    const int ko = (lane >> 4) * 8;
    for (int kk = 0; kk < C_; kk += 32) {
        v16h a, b;
        #pragma unroll
        for (int j = 0; j < 8; ++j) {
            a[j]     = (_Float16)prim[mrow * C_ + kk + ko + j];
            a[8 + j] = (_Float16)prim[mrow * C_ + kk + 16 + ko + j];
            b[j]     = (_Float16)prototype[nrow * C_ + kk + ko + j];
            b[8 + j] = (_Float16)prototype[nrow * C_ + kk + 16 + ko + j];
        }
        acc = __builtin_amdgcn_wmma_f32_16x16x32_f16(
                false, a, false, b, (short)0, acc, false, false);
    }
#endif

    // D layout: VGPR r -> M = r + 8*(lane>=16), N = lane%16. Apply 1/T = 5.
    #pragma unroll
    for (int r = 0; r < 8; ++r) {
        int M = (lane >> 4) * 8 + r;
        int Nc = lane & 15;
        s_sim[ti * 16 + M][tj * 16 + Nc] = acc[r] * 5.0f;
    }
    __syncthreads();

    // InfoNCE: nll_p = logsumexp_q(sim[p,q]) - sim[p,p]
    if (tid < 64) {
        float m = -3.4e38f;
        for (int q = 0; q < 64; ++q) m = fmaxf(m, s_sim[tid][q]);
        float sum = 0.0f;
        for (int q = 0; q < 64; ++q) sum += expf(s_sim[tid][q] - m);
        s_nll[tid] = m + logf(sum) - s_sim[tid][tid];
    }
    __syncthreads();
    if (tid == 0) {
        float t = 0.0f;
        for (int q = 0; q < 64; ++q) t += s_nll[q];
        out[0] = 0.1f * (t * (1.0f / 64.0f));   // W * mean
    }
}

// ---------------------------------------------------------------------------
extern "C" void kernel_launch(void* const* d_in, const int* in_sizes, int n_in,
                              void* d_out, int out_size, void* d_ws, size_t ws_size,
                              hipStream_t stream) {
    const int*   primlabel = (const int*)d_in[0];    // [BS,P,K]
    const float* features  = (const float*)d_in[1];  // [N,BS,C]
    const float* prototype = (const float*)d_in[2];  // [P,C]
    float* out = (float*)d_out;

    float* ws        = (float*)d_ws;
    float* invnorm   = ws;                          // N*BS            = 320000
    float* sums_bp   = invnorm + (N_ * BS_);        // BS*P*C          = 131072
    float* counts_bp = sums_bp + (BS_ * P_ * C_);   // BS*P            = 512
    float* prim      = counts_bp + (BS_ * P_);      // P*C             = 16384

    k_invnorm  <<<(N_ * BS_) / 8, 256, 0, stream>>>(features, invnorm);
    k_gather   <<<BS_ * P_,       256, 0, stream>>>(primlabel, features, invnorm,
                                                    sums_bp, counts_bp);
    k_mean_norm<<<P_,             256, 0, stream>>>(sums_bp, counts_bp, prototype, prim);
    k_sim_loss <<<1,              512, 0, stream>>>(prim, prototype, out);
}
